// OptimizedAttention_43327630082175
// MI455X (gfx1250) — compile-verified, matrix operand order checked by reading, not run on hardware
//
#include <hip/hip_runtime.h>
#include <hip/hip_bf16.h>

// ---------------------------------------------------------------------------
// Causal MHA forward for MI455X (gfx1250, wave32, WMMA 16x16x32 bf16).
// B=8 S=1024 E=1024 H=16 D=64.
//  - All GEMMs on v_wmma_f32_16x16x32_bf16, fp32 accumulation.
//  - Projection GEMMs stage the shared 64-col B tile per workgroup in LDS via
//    double-buffered global_load_async_to_lds_b128 (ASYNCcnt), padded to kill
//    bank conflicts; waves read B fragments with ds_load_b128.
//  - Flash attention computed transposed (S^T / O^T) so softmax stats are
//    per-lane and exp'd scores feed the P^T B-fragment with no shuffles.
// ---------------------------------------------------------------------------

typedef __attribute__((ext_vector_type(16))) __bf16 v16bf;
typedef __attribute__((ext_vector_type(8)))  __bf16 v8bf;
typedef __attribute__((ext_vector_type(8)))  float  v8f;

#define BB 8
#define SS 1024
#define EE 1024
#define HH 16
#define DD 64

// LDS B-stage: 64 rows (n) x 64 cols (k) bf16, row stride padded 64->72 elems
// (144B) so the 16 fragment rows hit 16 distinct 4-bank groups.
#define STG_ROWS   64
#define STG_K      64
#define STG_LDE    72                       // padded elements per row
#define STG_BYTES  (STG_ROWS * STG_LDE * 2) // 9216 B per buffer

__device__ __forceinline__ v8f zero8() {
  v8f z;
#pragma unroll
  for (int i = 0; i < 8; ++i) z[i] = 0.0f;
  return z;
}

__device__ __forceinline__ v8f wmma_bf16(v16bf a, v16bf b, v8f c) {
  return __builtin_amdgcn_wmma_f32_16x16x32_bf16(false, a, false, b,
                                                 (short)0, c, false, false);
}

// 16x32 bf16 operand fragment from row-major memory, leading dim `ld` elems.
// CDNA5 A-layout: lane {row=lane&15, half=lane>>4}; elem i<8 -> K=half*8+i,
// elem i>=8 -> K=16+half*8+(i-8).  Two b128 loads per lane.
__device__ __forceinline__ v16bf load_frag(const __bf16* p, int ld, int lane) {
  const int row  = lane & 15;
  const int half = lane >> 4;
  const __bf16* base = p + (size_t)row * ld + half * 8;
  v8bf lo = *(const v8bf*)(base);
  v8bf hi = *(const v8bf*)(base + 16);
  v16bf r;
#pragma unroll
  for (int i = 0; i < 8; ++i) { r[i] = lo[i]; r[i + 8] = hi[i]; }
  return r;
}

// Same fragment pattern out of the padded LDS stage (ds_load_b128 x2).
__device__ __forceinline__ v16bf load_frag_lds(const __bf16* sh, int row0,
                                               int k2, int lane) {
  const int row  = row0 + (lane & 15);
  const int half = lane >> 4;
  const __bf16* base = sh + (size_t)row * STG_LDE + k2 + half * 8;
  v8bf lo = *(const v8bf*)(base);
  v8bf hi = *(const v8bf*)(base + 16);
  v16bf r;
#pragma unroll
  for (int i = 0; i < 8; ++i) { r[i] = lo[i]; r[i + 8] = hi[i]; }
  return r;
}

// Issue the async fill of one 64x64 stage: 256 threads x 2 b128 each.
// Thread t covers 16B chunks of rows (t>>3) and 32+(t>>3), col bytes (t&7)*16.
__device__ __forceinline__ void stage_fill_async(const __bf16* BT, int n0,
                                                 int kk, int Kd,
                                                 const __bf16* shbuf, int tid) {
  const int row = tid >> 3;
  const int kb  = (tid & 7) * 16;  // byte offset within the 128B of row data
  unsigned long long g0 =
      (unsigned long long)(size_t)((const char*)(BT + (size_t)(n0 + row) * Kd + kk) + kb);
  unsigned long long g1 =
      (unsigned long long)(size_t)((const char*)(BT + (size_t)(n0 + 32 + row) * Kd + kk) + kb);
  unsigned l0 = (unsigned)(size_t)((const char*)(shbuf + (size_t)row * STG_LDE) + kb);
  unsigned l1 = (unsigned)(size_t)((const char*)(shbuf + (size_t)(32 + row) * STG_LDE) + kb);
  asm volatile("global_load_async_to_lds_b128 %0, %1, off"
               :: "v"(l0), "v"(g0) : "memory");
  asm volatile("global_load_async_to_lds_b128 %0, %1, off"
               :: "v"(l1), "v"(g1) : "memory");
}

__device__ __forceinline__ void wait_async_le2() {
  asm volatile("s_wait_asynccnt 0x2" ::: "memory");
}
__device__ __forceinline__ void wait_async_le0() {
  asm volatile("s_wait_asynccnt 0x0" ::: "memory");
}

// --------------------------- conversion kernels ----------------------------

__global__ void f32_to_bf16(const float* __restrict__ in,
                            __bf16* __restrict__ out, int n) {
  int i = blockIdx.x * blockDim.x + threadIdx.x;
  if (i < n) out[i] = (__bf16)in[i];
}

// in: fp32 [R, C] row-major -> out: bf16 [C, R] row-major (transposed)
__global__ void transpose_f32_to_bf16(const float* __restrict__ in,
                                      __bf16* __restrict__ out, int R, int C) {
  int i = blockIdx.x * blockDim.x + threadIdx.x;
  if (i >= R * C) return;
  int k = i % R;
  int n = i / R;
  out[(size_t)n * R + k] = (__bf16)in[(size_t)k * C + n];
}

// ------------------------------- QKV GEMM ----------------------------------
// Block = 8 waves = 256x64 macro tile (wave w -> rows m0+32w).  B tile (64 n
// x 64 k) staged in LDS per k-stage, double buffered via async-to-LDS.
// Epilogue: +bias, split into q (x 1/sqrt(D)) [B,H,S,D], k [B,H,S,D],
// v^T [B,H,D,S].
__global__ void gemm_qkv(const __bf16* __restrict__ A,
                         const __bf16* __restrict__ BT,
                         const float* __restrict__ bias,
                         __bf16* __restrict__ qo,
                         __bf16* __restrict__ ko,
                         __bf16* __restrict__ vto) {
  __shared__ __align__(16) __bf16 shB[2 * STG_ROWS * STG_LDE];
  const int M = BB * SS, N = 3 * EE, Kd = EE;
  const int NB = N / 64;
  const int tid  = threadIdx.x;
  const int lane = tid & 31;
  const int w    = tid >> 5;
  const int bm = blockIdx.x / NB;
  const int bn = blockIdx.x % NB;
  const int m0 = bm * 256 + w * 32;
  const int n0 = bn * 64;
  (void)M;

  v8f acc[2][4];
#pragma unroll
  for (int i = 0; i < 2; ++i)
#pragma unroll
    for (int g = 0; g < 4; ++g) acc[i][g] = zero8();

  stage_fill_async(BT, n0, 0, Kd, shB, tid);  // prime stage 0 -> buffer 0

  for (int ks = 0; ks < Kd; ks += STG_K) {
    const int buf = (ks >> 6) & 1;
    const __bf16* shbuf = shB + buf * STG_ROWS * STG_LDE;
    __syncthreads();  // prev compute done before next fill overwrites !buf
    if (ks + STG_K < Kd) {
      stage_fill_async(BT, n0, ks + STG_K, Kd,
                       shB + (buf ^ 1) * STG_ROWS * STG_LDE, tid);
      wait_async_le2();  // our stage-`buf` loads complete (2 newer in flight)
    } else {
      wait_async_le0();
    }
    __syncthreads();  // all waves' stage-`buf` loads complete

    const __bf16* Abase = A + (size_t)m0 * Kd + ks;
#pragma unroll
    for (int k2 = 0; k2 < STG_K; k2 += 32) {
      v16bf a0 = load_frag(Abase + k2, Kd, lane);
      v16bf a1 = load_frag(Abase + (size_t)16 * Kd + k2, Kd, lane);
#pragma unroll
      for (int g = 0; g < 4; ++g) {
        v16bf bf = load_frag_lds(shbuf, g * 16, k2, lane);
        acc[0][g] = wmma_bf16(a0, bf, acc[0][g]);
        acc[1][g] = wmma_bf16(a1, bf, acc[1][g]);
      }
    }
  }

  const int nl = lane & 15;
  const int mh = (lane >> 4) * 8;
#pragma unroll
  for (int ms = 0; ms < 2; ++ms) {
#pragma unroll
    for (int g = 0; g < 4; ++g) {
      int   col   = n0 + g * 16 + nl;
      int   which = col >> 10;  // 0=q 1=k 2=v (uniform per g group)
      int   e     = col & (EE - 1);
      int   h     = e >> 6, d = e & (DD - 1);
      float bv    = bias[col];
#pragma unroll
      for (int r = 0; r < 8; ++r) {
        int row = m0 + ms * 16 + mh + r;
        int bi  = row >> 10, s = row & (SS - 1);
        float v = acc[ms][g][r] + bv;
        size_t bh = (size_t)(bi * HH + h);
        if (which == 0)
          qo[(bh * SS + s) * DD + d] = (__bf16)(v * 0.125f);  // 1/sqrt(64)
        else if (which == 1)
          ko[(bh * SS + s) * DD + d] = (__bf16)v;
        else
          vto[(bh * DD + d) * SS + s] = (__bf16)v;            // transposed V
      }
    }
  }
}

// ---------------------------- Flash attention ------------------------------
__global__ void flash_attn(const __bf16* __restrict__ q,
                           const __bf16* __restrict__ k,
                           const __bf16* __restrict__ vt,
                           __bf16* __restrict__ ao) {
  const int QT = SS / 16;
  int wid  = (blockIdx.x * blockDim.x + threadIdx.x) >> 5;
  int lane = threadIdx.x & 31;
  if (wid >= BB * HH * QT) return;
  int qt = wid % QT;
  int bh = wid / QT;
  int h  = bh % HH, b = bh / HH;

  const __bf16* Q  = q  + (size_t)bh * SS * DD;   // [S, D] (pre-scaled)
  const __bf16* K  = k  + (size_t)bh * SS * DD;   // [S, D]
  const __bf16* VT = vt + (size_t)bh * DD * SS;   // [D, S]
  const int q0 = qt * 16;

  v16bf qf0 = load_frag(Q + (size_t)q0 * DD, DD, lane);
  v16bf qf1 = load_frag(Q + (size_t)q0 * DD + 32, DD, lane);

  v8f acc[4];  // O^T: lane holds (d = g*16+half8+r, q = lane&15)
#pragma unroll
  for (int g = 0; g < 4; ++g) acc[g] = zero8();

  float m_run = -1e30f, l_run = 0.0f;
  const int qg    = q0 + (lane & 15);
  const int half8 = (lane >> 4) * 8;

  for (int kvb = 0; kvb < q0 + 16; kvb += 32) {
    v8f st[2];
    st[0] = zero8(); st[1] = zero8();
#pragma unroll
    for (int j = 0; j < 2; ++j) {
      if (kvb + j * 16 < q0 + 16) {
        v16bf ka0 = load_frag(K + (size_t)(kvb + j * 16) * DD, DD, lane);
        v16bf ka1 = load_frag(K + (size_t)(kvb + j * 16) * DD + 32, DD, lane);
        st[j] = wmma_bf16(ka0, qf0, st[j]);
        st[j] = wmma_bf16(ka1, qf1, st[j]);
      }
    }
    float mx = -1e30f;
#pragma unroll
    for (int j = 0; j < 2; ++j)
#pragma unroll
      for (int r = 0; r < 8; ++r) {
        int   kvg = kvb + j * 16 + half8 + r;
        float s   = (kvg <= qg) ? st[j][r] : -1e30f;
        st[j][r]  = s;
        mx        = fmaxf(mx, s);
      }
    mx = fmaxf(mx, __shfl_xor(mx, 16, 32));
    float m_new = fmaxf(m_run, mx);
    float alpha = __expf(m_run - m_new);

    float  psum = 0.0f;
    __bf16 p0[8], p1[8];
#pragma unroll
    for (int r = 0; r < 8; ++r) {
      float e0 = __expf(st[0][r] - m_new);
      float e1 = __expf(st[1][r] - m_new);
      psum += e0 + e1;
      p0[r] = (__bf16)e0;
      p1[r] = (__bf16)e1;
    }
    psum += __shfl_xor(psum, 16, 32);
    l_run = l_run * alpha + psum;
    m_run = m_new;

    v16bf pb;  // P^T B-fragment straight from exp'd C-layout registers
#pragma unroll
    for (int r = 0; r < 8; ++r) { pb[r] = p0[r]; pb[r + 8] = p1[r]; }

#pragma unroll
    for (int g = 0; g < 4; ++g) {
      v16bf va = load_frag(VT + (size_t)(g * 16) * SS + kvb, SS, lane);
#pragma unroll
      for (int r = 0; r < 8; ++r) acc[g][r] *= alpha;
      acc[g] = wmma_bf16(va, pb, acc[g]);
    }
  }

  float inv = (l_run > 0.0f) ? (1.0f / l_run) : 0.0f;
  int   s   = q0 + (lane & 15);
#pragma unroll
  for (int g = 0; g < 4; ++g)
#pragma unroll
    for (int r = 0; r < 8; ++r) {
      int d = g * 16 + half8 + r;
      ao[((size_t)b * SS + s) * EE + h * DD + d] = (__bf16)(acc[g][r] * inv);
    }
}

// ----------------------------- output GEMM ---------------------------------
__global__ void gemm_proj(const __bf16* __restrict__ A,
                          const __bf16* __restrict__ BT,
                          const float* __restrict__ bias,
                          float* __restrict__ out) {
  __shared__ __align__(16) __bf16 shB[2 * STG_ROWS * STG_LDE];
  const int N = EE, Kd = EE;
  const int NB = N / 64;
  const int tid  = threadIdx.x;
  const int lane = tid & 31;
  const int w    = tid >> 5;
  const int bm = blockIdx.x / NB;
  const int bn = blockIdx.x % NB;
  const int m0 = bm * 256 + w * 32;
  const int n0 = bn * 64;

  v8f acc[2][4];
#pragma unroll
  for (int i = 0; i < 2; ++i)
#pragma unroll
    for (int g = 0; g < 4; ++g) acc[i][g] = zero8();

  stage_fill_async(BT, n0, 0, Kd, shB, tid);

  for (int ks = 0; ks < Kd; ks += STG_K) {
    const int buf = (ks >> 6) & 1;
    const __bf16* shbuf = shB + buf * STG_ROWS * STG_LDE;
    __syncthreads();
    if (ks + STG_K < Kd) {
      stage_fill_async(BT, n0, ks + STG_K, Kd,
                       shB + (buf ^ 1) * STG_ROWS * STG_LDE, tid);
      wait_async_le2();
    } else {
      wait_async_le0();
    }
    __syncthreads();

    const __bf16* Abase = A + (size_t)m0 * Kd + ks;
#pragma unroll
    for (int k2 = 0; k2 < STG_K; k2 += 32) {
      v16bf a0 = load_frag(Abase + k2, Kd, lane);
      v16bf a1 = load_frag(Abase + (size_t)16 * Kd + k2, Kd, lane);
#pragma unroll
      for (int g = 0; g < 4; ++g) {
        v16bf bf = load_frag_lds(shbuf, g * 16, k2, lane);
        acc[0][g] = wmma_bf16(a0, bf, acc[0][g]);
        acc[1][g] = wmma_bf16(a1, bf, acc[1][g]);
      }
    }
  }

  const int nl = lane & 15;
  const int mh = (lane >> 4) * 8;
#pragma unroll
  for (int ms = 0; ms < 2; ++ms)
#pragma unroll
    for (int g = 0; g < 4; ++g) {
      int   col = n0 + g * 16 + nl;
      float bv  = bias[col];
#pragma unroll
      for (int r = 0; r < 8; ++r) {
        int row = m0 + ms * 16 + mh + r;
        out[(size_t)row * N + col] = acc[ms][g][r] + bv;
      }
    }
}

// ------------------------------- launcher ----------------------------------

extern "C" void kernel_launch(void* const* d_in, const int* in_sizes, int n_in,
                              void* d_out, int out_size, void* d_ws,
                              size_t ws_size, hipStream_t stream) {
  const float* x  = (const float*)d_in[0];  // [8, 1024, 1024]
  const float* Wa = (const float*)d_in[1];  // [1024, 3072]
  const float* ba = (const float*)d_in[2];  // [3072]
  const float* Wp = (const float*)d_in[3];  // [1024, 1024]
  const float* bp = (const float*)d_in[4];  // [1024]
  float* out = (float*)d_out;               // [8, 1024, 1024]

  // Workspace layout: 88 MB of bf16 intermediates (fits in 192 MB L2).
  char* ws = (char*)d_ws;
  __bf16* xb   = (__bf16*)(ws + ((size_t)0  << 20));  // x bf16        16 MB
  __bf16* wat  = (__bf16*)(ws + ((size_t)16 << 20));  // W_attn^T       6 MB
  __bf16* wpt  = (__bf16*)(ws + ((size_t)22 << 20));  // W_proj^T       2 MB
  __bf16* qb   = (__bf16*)(ws + ((size_t)24 << 20));  // q [B,H,S,D]   16 MB
  __bf16* kbuf = (__bf16*)(ws + ((size_t)40 << 20));  // k [B,H,S,D]   16 MB
  __bf16* vtb  = (__bf16*)(ws + ((size_t)56 << 20));  // v^T [B,H,D,S] 16 MB
  __bf16* ao   = (__bf16*)(ws + ((size_t)72 << 20));  // attn [B,S,E]  16 MB

  const int nX = BB * SS * EE;  // 8388608
  f32_to_bf16<<<(nX + 255) / 256, 256, 0, stream>>>(x, xb, nX);
  transpose_f32_to_bf16<<<(EE * 3 * EE + 255) / 256, 256, 0, stream>>>(
      Wa, wat, EE, 3 * EE);
  transpose_f32_to_bf16<<<(EE * EE + 255) / 256, 256, 0, stream>>>(
      Wp, wpt, EE, EE);

  const int qkvBlocks = (BB * SS / 256) * (3 * EE / 64);  // 32*48 = 1536
  gemm_qkv<<<qkvBlocks, 256, 0, stream>>>(xb, wat, ba, qb, kbuf, vtb);

  const int faWaves = BB * HH * (SS / 16);  // 8192
  flash_attn<<<faWaves / 8, 256, 0, stream>>>(qb, kbuf, vtb, ao);

  const int projBlocks = (BB * SS / 256) * (EE / 64);  // 32*16 = 512
  gemm_proj<<<projBlocks, 256, 0, stream>>>(ao, wpt, bp, out);
}
